// AttentionGatedMSG_25958782337262
// MI455X (gfx1250) — matrix operand order repaired and checked by reading, not compiled, not run on hardware
//
#include <hip/hip_runtime.h>

// ---------------- CDNA5 (gfx1250) wave32 WMMA implementation -----------------
// Shapes: B=4, C=512, H=W=64, NPIX=16384.
// All dense contractions run on v_wmma_f32_16x16x32_bf16.
// All 12 independent 9-channel convs fused into ONE launch (grid 64x4x3,
// 3072 waves in flight) for WGP occupancy; dw halves summed in dyn_att.
// Conv: 4 N-tiles per wave (A-fragment reuse). Comb GEMM: 4 M-tiles per wave
// (B-fragment reuse). Boundary handling via branch-free zero-page selects.

#define Bsz 4
#define Cc  512
#define Hh  64
#define Ww  64
#define NPIX (Bsz*Hh*Ww)            // 16384
#define KB_CONV 144                  // (9 taps * 512 ch) / 32
#define KB_COMB 48                   // 1536 / 32
#define PACK_CONV_DW (KB_CONV*32*8)  // dwords per packed conv weight set = 36864
#define PACK_COMB_DW (32*KB_COMB*32*8)

typedef __attribute__((ext_vector_type(16))) __bf16       bf16x16;
typedef __attribute__((ext_vector_type(8)))  float        floatx8;
typedef __attribute__((ext_vector_type(8)))  unsigned int uintx8;

__device__ __forceinline__ unsigned short f2bf(float f) {
  unsigned int u = __builtin_bit_cast(unsigned int, f);
  unsigned int r = u + 0x7FFFu + ((u >> 16) & 1u);   // round-to-nearest-even
  return (unsigned short)(r >> 16);
}
__device__ __forceinline__ float bf2f(unsigned short s) {
  unsigned int u = ((unsigned int)s) << 16;
  return __builtin_bit_cast(float, u);
}
__device__ __forceinline__ unsigned int hashu(unsigned int x) {
  x ^= x >> 16; x *= 0x7feb352dU; x ^= x >> 15; x *= 0x846ca68bU; x ^= x >> 16;
  return x;
}
// Deterministic stand-in for jax.random.normal(fold_in(key(42), fold), ...)
__device__ __forceinline__ float rnormal(int fold, int idx) {
  unsigned h1 = hashu(0x9E3779B9u * (unsigned)fold + 0x85EBCA6Bu * (unsigned)idx + 0x42u);
  unsigned h2 = hashu(h1 + 0x27220A95u);
  float u1 = ((h1 >> 8) + 1u) * (1.0f / 16777217.0f);
  float u2 = (h2 >> 8) * (1.0f / 16777216.0f);
  return sqrtf(-2.0f * __logf(u1)) * __cosf(6.28318530718f * u2);
}

// ---- zero page for boundary-tap loads --------------------------------------
__global__ void zfill(unsigned short* __restrict__ zb) {
  int t = threadIdx.x;
  for (int i = t; i < 1024; i += 256) zb[i] = 0;
}

// ---- NCHW f32 -> NHWC bf16 --------------------------------------------------
__global__ void cvt_nhwc(const float* __restrict__ src, unsigned short* __restrict__ dst) {
  int e = blockIdx.x * 256 + threadIdx.x;       // over NPIX*Cc
  int c = e & (Cc - 1);
  int pix = e >> 9;
  int b = pix >> 12, hw = pix & 4095;
  dst[e] = f2bf(src[((size_t)(b * Cc + c)) * 4096 + hw]);
}

// ---- Pack OIHW f32 weights into WMMA A-fragment order (bf16) ---------------
// A fragment, contiguous k order chosen here (B matches): lane l -> row m=l%16; K base = (l<16?0:16);
// dwords 0..7 = K pairs {base+0..15}.
// K ordering: k = tap*Cin + c.
__global__ void pack_w(const float* __restrict__ w, int Mrows, int CinTot, int c_off,
                       int Cin, int ntaps, int KB, unsigned int* __restrict__ out) {
  int D = blockIdx.x * 256 + threadIdx.x;
  int within = D & 7;
  int lane   = (D >> 3) & 31;
  int kb     = (D >> 8) % KB;
  int mt     = (D >> 8) / KB;
  int m = mt * 16 + (lane & 15);
  int kbase = kb * 32 + ((lane & 16) ? 16 : 0) + within * 2;
  unsigned int dw = 0;
  for (int t = 0; t < 2; ++t) {
    int k = kbase + t;
    int tap = k / Cin, c = k % Cin;
    unsigned short hv = 0;
    if (m < Mrows && tap < ntaps)
      hv = f2bf(w[((size_t)m * CinTot + c_off + c) * ntaps + tap]);
    dw |= ((unsigned int)hv) << (16 * t);
  }
  out[D] = dw;
}

// ---- ALL 12 9-out-channel dilated 3x3 convs as one fused WMMA GEMM launch --
// grid (64, set=0..3, branch=0..2). Per branch: set0=kp(se half), set1=kp(sr
// half), set2=kse(se), set3=ksr(sr). One wave per FOUR 16-pixel N-tiles
// (A reused 4x). Tap geometry hoisted; boundary taps -> zero page.
struct ConvArgs {
  const unsigned int* wp;        // 12 packed weight sets
  const unsigned short* se;      // NHWC bf16
  const unsigned short* sr;      // NHWC bf16
  const unsigned short* zb;      // zero page
  const float* bias[12];         // per (branch*4+set); null => no bias
  float* out;                    // 12 * NPIX*16 f32 slices
};
__global__ void conv9_wmma(ConvArgs a) {
  int set = blockIdx.y, branch = blockIdx.z;
  int slot = branch * 4 + set;
  int dil = (branch == 0) ? 1 : ((branch == 1) ? 4 : 8);
  const unsigned int* wpack = a.wp + (size_t)slot * PACK_CONV_DW;
  const unsigned short* x = (set & 1) ? a.sr : a.se;
  const float* bias9 = a.bias[slot];
  float* out16 = a.out + (size_t)slot * NPIX * 16;

  int wave = (blockIdx.x * blockDim.x + threadIdx.x) >> 5;   // 0..255 (exact)
  int lane = threadIdx.x & 31;
  int chalf = (lane & 16) ? 16 : 0;
  int p0 = wave * 64 + (lane & 15);           // tile j pixel = p0 + j*16
  int hh[4], ww[4], bb[4];
#pragma unroll
  for (int j = 0; j < 4; ++j) {
    int p = p0 + j * 16;
    bb[j] = p >> 12; int hw = p & 4095; hh[j] = hw >> 6; ww[j] = hw & 63;
  }
  floatx8 acc[4] = {{}, {}, {}, {}};
  for (int tap = 0; tap < 9; ++tap) {
    int di = (tap >= 6) ? 2 : ((tap >= 3) ? 1 : 0);
    int dj = tap - di * 3;
    int dh = (di - 1) * dil, dw2 = (dj - 1) * dil;
    const unsigned short* bp[4];
#pragma unroll
    for (int j = 0; j < 4; ++j) {
      int h2 = hh[j] + dh, w2 = ww[j] + dw2;
      bool ok = ((unsigned)h2 < 64u) && ((unsigned)w2 < 64u);
      size_t ad = ((size_t)((((bb[j] << 6) + h2) << 6) + w2)) * Cc + chalf;
      bp[j] = ok ? (x + ad) : (a.zb + chalf);
    }
    const unsigned int* ap = wpack + ((size_t)tap * 16 * 32 + lane) * 8;
#pragma unroll 4
    for (int kc = 0; kc < 16; ++kc) {
      bf16x16 af = __builtin_bit_cast(bf16x16, *(const uintx8*)(ap + (size_t)kc * 256));
#pragma unroll
      for (int j = 0; j < 4; ++j) {
        bf16x16 bfr = __builtin_bit_cast(bf16x16, *(const uintx8*)(bp[j] + kc * 32));
        acc[j] = __builtin_amdgcn_wmma_f32_16x16x32_bf16(
            false, af, false, bfr, (short)0, acc[j], false, false);
      }
    }
  }
  int mbase = (lane & 16) ? 8 : 0;
#pragma unroll
  for (int j = 0; j < 4; ++j) {
    int p = p0 + j * 16;
    float* op = out16 + ((size_t)p << 4) + mbase;
    floatx8 r = acc[j];
    if (bias9) {
#pragma unroll
      for (int v = 0; v < 8; ++v) { int mm = mbase + v; r[v] += (mm < 9) ? bias9[mm] : 0.f; }
    }
    *(floatx8*)op = r;
  }
}

// ---- se_term / att assembly (weighted sum of 9 dilated shifts) -------------
// dw = dwA + dwB (the two halves of the kp conv over cat=[se;sr]).
__global__ void dyn_att(const unsigned short* __restrict__ se_bf,
                        const unsigned short* __restrict__ sr_bf,
                        const float* __restrict__ dwA, const float* __restrict__ dwB,
                        const float* __restrict__ kse16, const float* __restrict__ ksr16,
                        unsigned short* __restrict__ se_term_bf,
                        float* __restrict__ att, int dil) {
  int e = blockIdx.x * 256 + threadIdx.x;       // over NPIX*Cc
  int c = e & (Cc - 1);
  int pix = e >> 9;
  int b = pix >> 12, hw = pix & 4095, h = hw >> 6, w = hw & 63;
  const float* dwa  = dwA   + ((size_t)pix << 4);
  const float* dwb  = dwB   + ((size_t)pix << 4);
  const float* ksep = kse16 + ((size_t)pix << 4);
  const float* ksrp = ksr16 + ((size_t)pix << 4);
  float a_dw = 0.f, a_se = 0.f, a_sr = 0.f;
#pragma unroll
  for (int k = 0; k < 9; ++k) {
    int h2 = h + (k / 3 - 1) * dil;
    int w2 = w + (k % 3 - 1) * dil;
    float sev = 0.f, srv = 0.f;
    if ((unsigned)h2 < 64u && (unsigned)w2 < 64u) {
      size_t ad = ((size_t)((((b << 6) + h2) << 6) + w2)) * Cc + c;
      sev = bf2f(se_bf[ad]); srv = bf2f(sr_bf[ad]);
    }
    a_dw += (dwa[k] + dwb[k]) * sev;
    a_se += ksep[k] * sev;
    a_sr += ksrp[k] * srv;
  }
  float srC = bf2f(sr_bf[(size_t)pix * Cc + c]);
  se_term_bf[e] = f2bf(a_dw);
  att[e] = srC * a_dw + a_se + a_sr;
}

// ---- per-(i, batch) channel softmax weights --------------------------------
__global__ void chag_k(float* __restrict__ chag, const int* __restrict__ rankp, int branch) {
  int b = blockIdx.x, i = blockIdx.y;
  int R = *rankp; if (R > 8) R = 8;
  if (i >= R) return;
  __shared__ float red[256];
  int t = threadIdx.x;
  float v0 = rnormal(branch * 100 + i, b * Cc + t);
  float v1 = rnormal(branch * 100 + i, b * Cc + t + 256);
  red[t] = fmaxf(v0, v1); __syncthreads();
  for (int s = 128; s > 0; s >>= 1) { if (t < s) red[t] = fmaxf(red[t], red[t + s]); __syncthreads(); }
  float mx = red[0]; __syncthreads();
  float e0 = __expf(v0 - mx), e1 = __expf(v1 - mx);
  red[t] = e0 + e1; __syncthreads();
  for (int s = 128; s > 0; s >>= 1) { if (t < s) red[t] += red[t + s]; __syncthreads(); }
  float inv = 1.f / red[0];
  float* o = chag + ((size_t)i * Bsz + b) * Cc;
  o[t] = e0 * inv; o[t + 256] = e1 * inv;
}

// ---- structured attention + msg write (and att1 output for branch 0) -------
__global__ void struc_k(const float* __restrict__ att,
                        const unsigned short* __restrict__ se_term_bf,
                        const float* __restrict__ chag, const int* __restrict__ rankp,
                        unsigned short* __restrict__ m_bf, int branch,
                        float* __restrict__ att1_out) {
  int pix = blockIdx.x;
  int b = pix >> 12, hw = pix & 4095;
  int t = threadIdx.x;
  int R = *rankp; if (R > 8) R = 8;
  const float* ap = att + ((size_t)pix << 9);
  float a0 = ap[t], a1 = ap[t + 256];
  __shared__ float red[256];
  __shared__ float sig[8];
  for (int i = 0; i < R; ++i) {
    const float* cg = chag + ((size_t)i * Bsz + b) * Cc;
    red[t] = cg[t] * a0 + cg[t + 256] * a1;
    __syncthreads();
    for (int s = 128; s > 0; s >>= 1) { if (t < s) red[t] += red[t + s]; __syncthreads(); }
    if (t == 0) sig[i] = 1.f / (1.f + __expf(-red[0]));
    __syncthreads();
  }
  float o0, o1;
  if (R == 0) { o0 = a0; o1 = a1; }
  else {
    o0 = 0.f; o1 = 0.f;
    for (int i = 0; i < R; ++i) {
      const float* cg = chag + ((size_t)i * Bsz + b) * Cc;
      o0 += sig[i] * cg[t] * a0;
      o1 += sig[i] * cg[t + 256] * a1;
    }
  }
  size_t mb = (size_t)pix * (3 * Cc) + branch * Cc;
  m_bf[mb + t]       = f2bf(o0 * bf2f(se_term_bf[((size_t)pix << 9) + t]));
  m_bf[mb + t + 256] = f2bf(o1 * bf2f(se_term_bf[((size_t)pix << 9) + t + 256]));
  if (att1_out) {
    att1_out[((size_t)(b * Cc + t)) * 4096 + hw]       = o0;
    att1_out[((size_t)(b * Cc + t + 256)) * 4096 + hw] = o1;
  }
}

// ---- final 1x1 conv (M=512, K=1536, N=16384) WMMA GEMM + bias + ReLU -------
// One wave per FOUR M-tiles (B fragment reused 4x).
__global__ void comb_wmma(const unsigned int* __restrict__ wpack,
                          const unsigned short* __restrict__ m_bf,
                          const float* __restrict__ bias,
                          float* __restrict__ out) {
  int gw = (blockIdx.x * blockDim.x + threadIdx.x) >> 5;   // 0..8191 (exact)
  int lane = threadIdx.x & 31;
  int mg = gw >> 10;            // 0..7 : group of 4 out-channel tiles
  int nt = gw & 1023;           // pixel tile
  int p = nt * 16 + (lane & 15);
  int b = p >> 12, hw = p & 4095;
  int chalf = (lane & 16) ? 16 : 0;
  floatx8 acc[4] = {{}, {}, {}, {}};
  const unsigned short* bp = m_bf + (size_t)p * (3 * Cc) + chalf;
  const unsigned int* ap0 = wpack + (((size_t)(mg * 4) * KB_COMB) * 32 + lane) * 8;
#pragma unroll 4
  for (int kb = 0; kb < KB_COMB; ++kb) {
    bf16x16 bfr = __builtin_bit_cast(bf16x16, *(const uintx8*)(bp + kb * 32));
#pragma unroll
    for (int j = 0; j < 4; ++j) {
      bf16x16 af = __builtin_bit_cast(bf16x16,
          *(const uintx8*)(ap0 + ((size_t)j * KB_COMB + kb) * 256));
      acc[j] = __builtin_amdgcn_wmma_f32_16x16x32_bf16(
          false, af, false, bfr, (short)0, acc[j], false, false);
    }
  }
  int mbase = (lane & 16) ? 8 : 0;
#pragma unroll
  for (int j = 0; j < 4; ++j) {
    int mt = mg * 4 + j;
#pragma unroll
    for (int v = 0; v < 8; ++v) {
      int o = mt * 16 + mbase + v;
      float val = fmaxf(acc[j][v] + bias[o], 0.f);
      out[((size_t)(b * Cc + o)) * 4096 + hw] = val;
    }
  }
}

// ----------------------------------------------------------------------------
extern "C" void kernel_launch(void* const* d_in, const int* in_sizes, int n_in,
                              void* d_out, int out_size, void* d_ws, size_t ws_size,
                              hipStream_t stream) {
  const float* sr = (const float*)d_in[0];
  const float* se = (const float*)d_in[1];
  const float* kpw[3]  = {(const float*)d_in[2],  (const float*)d_in[4],  (const float*)d_in[6]};
  const float* kpb[3]  = {(const float*)d_in[3],  (const float*)d_in[5],  (const float*)d_in[7]};
  const float* ksew[3] = {(const float*)d_in[8],  (const float*)d_in[12], (const float*)d_in[16]};
  const float* kseb[3] = {(const float*)d_in[9],  (const float*)d_in[13], (const float*)d_in[17]};
  const float* ksrw[3] = {(const float*)d_in[10], (const float*)d_in[14], (const float*)d_in[18]};
  const float* ksrb[3] = {(const float*)d_in[11], (const float*)d_in[15], (const float*)d_in[19]};
  const float* combw = (const float*)d_in[20];
  const float* combb = (const float*)d_in[21];
  const int*   rankp = (const int*)d_in[23];
  (void)in_sizes; (void)n_in; (void)out_size; (void)ws_size;

  char* ws = (char*)d_ws;
  size_t off = 0;
  auto alloc = [&](size_t bytes) -> void* {
    void* pp = ws + off;
    off = (off + bytes + 255) & ~(size_t)255;
    return pp;
  };
  unsigned short* se_bf   = (unsigned short*)alloc((size_t)NPIX * Cc * 2);
  unsigned short* sr_bf   = (unsigned short*)alloc((size_t)NPIX * Cc * 2);
  unsigned int*   wp_conv = (unsigned int*)  alloc((size_t)12 * PACK_CONV_DW * 4);
  unsigned int*   wp_comb = (unsigned int*)  alloc((size_t)PACK_COMB_DW * 4);
  float*          conv16  = (float*)         alloc((size_t)12 * NPIX * 16 * 4);
  unsigned short* set_bf  = (unsigned short*)alloc((size_t)NPIX * Cc * 2);
  float*          attb    = (float*)         alloc((size_t)NPIX * Cc * 4);
  float*          chag    = (float*)         alloc((size_t)8 * Bsz * Cc * 4);
  unsigned short* m_bf    = (unsigned short*)alloc((size_t)NPIX * 3 * Cc * 2);
  unsigned short* zb      = (unsigned short*)alloc((size_t)1024 * 2);

  // zero page + layout conversion
  zfill<<<1, 256, 0, stream>>>(zb);
  cvt_nhwc<<<32768, 256, 0, stream>>>(se, se_bf);
  cvt_nhwc<<<32768, 256, 0, stream>>>(sr, sr_bf);

  // weight packing: per branch {kp(se half), kp(sr half), kse, ksr}
  for (int bi = 0; bi < 3; ++bi) {
    unsigned int* base = wp_conv + (size_t)bi * 4 * PACK_CONV_DW;
    pack_w<<<144, 256, 0, stream>>>(kpw[bi],  9, 1024,   0, 512, 9, KB_CONV, base + 0 * PACK_CONV_DW);
    pack_w<<<144, 256, 0, stream>>>(kpw[bi],  9, 1024, 512, 512, 9, KB_CONV, base + 1 * PACK_CONV_DW);
    pack_w<<<144, 256, 0, stream>>>(ksew[bi], 9,  512,   0, 512, 9, KB_CONV, base + 2 * PACK_CONV_DW);
    pack_w<<<144, 256, 0, stream>>>(ksrw[bi], 9,  512,   0, 512, 9, KB_CONV, base + 3 * PACK_CONV_DW);
  }
  pack_w<<<1536, 256, 0, stream>>>(combw, 512, 1536, 0, 1536, 1, KB_COMB, wp_comb);

  // all 12 convs in one launch: 768 blocks, 3072 waves in flight
  ConvArgs ca;
  ca.wp = wp_conv; ca.se = se_bf; ca.sr = sr_bf; ca.zb = zb; ca.out = conv16;
  for (int bi = 0; bi < 3; ++bi) {
    ca.bias[bi * 4 + 0] = kpb[bi];
    ca.bias[bi * 4 + 1] = nullptr;      // sr-half of kp conv: bias already in half 0
    ca.bias[bi * 4 + 2] = kseb[bi];
    ca.bias[bi * 4 + 3] = ksrb[bi];
  }
  conv9_wmma<<<dim3(64, 4, 3), 128, 0, stream>>>(ca);

  const int dils[3] = {1, 4, 8};
  for (int bi = 0; bi < 3; ++bi) {
    float* cbase = conv16 + (size_t)bi * 4 * NPIX * 16;
    dyn_att<<<32768, 256, 0, stream>>>(se_bf, sr_bf,
                                       cbase + 0 * (size_t)NPIX * 16,   // dw (se half)
                                       cbase + 1 * (size_t)NPIX * 16,   // dw (sr half)
                                       cbase + 2 * (size_t)NPIX * 16,   // kse
                                       cbase + 3 * (size_t)NPIX * 16,   // ksr
                                       set_bf, attb, dils[bi]);
    chag_k<<<dim3(4, 8), 256, 0, stream>>>(chag, rankp, bi);
    struc_k<<<16384, 256, 0, stream>>>(attb, set_bf, chag, rankp, m_bf, bi,
                                       (bi == 0) ? ((float*)d_out + (size_t)Bsz * Cc * 4096) : nullptr);
  }
  // message_f = relu(comb(m)) -> d_out[0 .. B*C*H*W)
  comb_wmma<<<2048, 128, 0, stream>>>(wp_comb, m_bf, combb, (float*)d_out);
}
